// Net_19885698580767
// MI455X (gfx1250) — matrix operand order repaired and checked by reading, not compiled, not run on hardware
//
#include <hip/hip_runtime.h>
#include <hip/hip_bf16.h>
#include <math.h>

typedef __attribute__((ext_vector_type(16))) _Float16 v16h;
typedef __attribute__((ext_vector_type(8)))  _Float16 v8h;
typedef __attribute__((ext_vector_type(8)))  float    v8f;

__device__ __forceinline__ float eluf(float x) { return x > 0.f ? x : expm1f(x); }

// ---------------------------------------------------------------------------
// Encoder: out = elu(elu(x@w1+b1)@w2+b2), per-node scalar (tiny fan-ins).
// ---------------------------------------------------------------------------
__global__ void enc_kernel(const float* __restrict__ x, int nin,
                           const float* __restrict__ w1, const float* __restrict__ b1,
                           const float* __restrict__ w2, const float* __restrict__ b2,
                           float* __restrict__ out, int N) {
    __shared__ float sw1[6 * 16], sb1[16], sw2[256], sb2[16];
    int tid = threadIdx.x;
    for (int i = tid; i < nin * 16; i += blockDim.x) sw1[i] = w1[i];
    for (int i = tid; i < 256; i += blockDim.x) sw2[i] = w2[i];
    if (tid < 16) { sb1[tid] = b1[tid]; sb2[tid] = b2[tid]; }
    __syncthreads();
    int node = blockIdx.x * blockDim.x + tid;
    if (node >= N) return;
    float xin[6];
    for (int k = 0; k < nin; ++k) xin[k] = x[node * nin + k];
    float h1[16];
    for (int n = 0; n < 16; ++n) {
        float s = sb1[n];
        for (int k = 0; k < nin; ++k) s += xin[k] * sw1[k * 16 + n];
        h1[n] = eluf(s);
    }
    float* op = out + node * 16;
    for (int n = 0; n < 16; ++n) {
        float s = sb2[n];
        #pragma unroll
        for (int k = 0; k < 16; ++k) s += h1[k] * sw2[k * 16 + n];
        op[n] = eluf(s);
    }
}

// ---------------------------------------------------------------------------
// Prep: per-16-node tile via WMMA.
//   mode 0: out = X @ W_bot            (U),  also writes nrm = |x|^2
//   mode 1: out = X @ (W_top - W_bot) + conv_b   (C)
// Also (if x16 != null) emits a packed f16 copy of X (node-major, 16 halves)
// so the kNN kernel can build WMMA fragments with plain b128 vector loads.
// conv_w is (32,16) row-major; W_top = rows 0..15, W_bot = rows 16..31.
// ---------------------------------------------------------------------------
__global__ void prep_kernel(const float* __restrict__ X,
                            const float* __restrict__ cw, const float* __restrict__ cb,
                            float* __restrict__ out, float* __restrict__ nrm,
                            _Float16* __restrict__ x16, int mode) {
    int lane = threadIdx.x;                  // blockDim == 32
    int tileBase = blockIdx.x * 16;
    int n = lane & 15;
    // A: 16 nodes x 16 features (K padded to 32 with zeros); lanes 0-15 K=0..15, lanes 16-31 pad
    v16h a;
    {
        const float* p = X + (tileBase + n) * 16;
        #pragma unroll
        for (int i = 0; i < 16; ++i) a[i] = (lane < 16) ? (_Float16)p[i] : (_Float16)0.f;
    }
    // B: 32x16 weights; lanes 0-15 hold K=0..15 of column `lane`, lanes 16-31 zero pad
    v16h b;
    if (lane < 16) {
        #pragma unroll
        for (int i = 0; i < 16; ++i) {
            float w = cw[(i + 16) * 16 + n];                 // W_bot
            if (mode) w = cw[i * 16 + n] - w;                // W_top - W_bot
            b[i] = (_Float16)w;
        }
    } else {
        #pragma unroll
        for (int i = 0; i < 16; ++i) b[i] = (_Float16)0.f;  // K = 16..31 pad
    }
    v8f c = {};
    c = __builtin_amdgcn_wmma_f32_16x16x32_f16(false, a, false, b, (short)0, c, false, false);
    float bias = mode ? cb[n] : 0.f;
    int hi = (lane < 16) ? 0 : 8;
    #pragma unroll
    for (int r = 0; r < 8; ++r)
        out[(tileBase + r + hi) * 16 + n] = c[r] + bias;
    if (lane < 16) {
        const float* q = X + (tileBase + lane) * 16;
        float qv[16];
        #pragma unroll
        for (int d = 0; d < 16; ++d) qv[d] = q[d];
        if (x16) {
            _Float16* o = x16 + (tileBase + lane) * 16;
            #pragma unroll
            for (int d = 0; d < 16; ++d) o[d] = (_Float16)qv[d];
        }
        if (mode == 0) {
            float s = 0.f;
            #pragma unroll
            for (int d = 0; d < 16; ++d) s += qv[d] * qv[d];
            nrm[tileBase + lane] = s;
        }
    }
}

// ---------------------------------------------------------------------------
// kNN + aggregate: one wave per 16-dst-row tile. Scores via WMMA (Xd·Xs^T),
// ranked by (|xs|^2 - 2*dot) == d2 up to a per-row constant. Top-16 per row,
// then feats = max_t elu(C_i + U_j).
// Features come in pre-packed f16 (node-major). srcPerGraph is always 192.
// ---------------------------------------------------------------------------
#define SCST 196   // LDS row stride (196 % 64 == 4 -> conflict-free for 16 lanes)

__global__ void knn_kernel(const _Float16* __restrict__ Xd16,
                           const _Float16* __restrict__ Xs16,
                           const float* __restrict__ U,  const float* __restrict__ C,
                           const float* __restrict__ nrm, float* __restrict__ out,
                           int dstPerGraph, int tilesPerGraph) {
    __shared__ float sc[16 * SCST];
    int lane = threadIdx.x;                  // blockDim == 32
    int tile = blockIdx.x;
    int g = tile / tilesPerGraph;
    int dstBase = g * dstPerGraph + (tile % tilesPerGraph) * 16;
    int srcBase = g * 192;
    int n = lane & 15;
    int hi = (lane < 16) ? 0 : 8;

    // Prefetch this graph's U slice (192 rows * 64B = 96 cachelines): the
    // data-dependent gather below will then hit WGP$/L2.
    {
        const char* ubase = (const char*)(U + (size_t)srcBase * 16);
        #pragma unroll
        for (int i = 0; i < 3; ++i)
            __builtin_prefetch(ubase + (size_t)(lane + 32 * i) * 128, 0, 3);
    }

    // A: dst tile features; lanes 0-15 load the full 16-half packed row (K=0..15),
    // lanes 16-31 are the K=16..31 zero pad.
    v16h a;
    if (lane < 16) {
        a = *(const v16h*)(Xd16 + (size_t)(dstBase + n) * 16);
    } else {
        #pragma unroll
        for (int i = 0; i < 16; ++i) a[i] = (_Float16)0.f;
    }

    for (int chunk = 0; chunk < 12; ++chunk) {
        // B: lanes 0-15 hold the full 16-half feature vector of src node
        // (chunk*16+lane); lanes 16-31 are the K=16..31 zero pad.
        v16h b;
        if (lane < 16) {
            b = *(const v16h*)(Xs16 + (size_t)(srcBase + chunk * 16 + n) * 16);
        } else {
            #pragma unroll
            for (int i = 0; i < 16; ++i) b[i] = (_Float16)0.f;
        }
        v8f c = {};
        c = __builtin_amdgcn_wmma_f32_16x16x32_f16(false, a, false, b, (short)0, c, false, false);
        float nr = nrm[srcBase + chunk * 16 + n];
        #pragma unroll
        for (int r = 0; r < 8; ++r)
            sc[(r + hi) * SCST + chunk * 16 + n] = nr - 2.0f * c[r];
    }
    __syncthreads();

    // Top-16 extraction: lane pair (row, row+16) splits the 192 candidates 96/96.
    int row = n;
    int grp = lane >> 4;
    int jlo = grp * 96;
    float* srow = sc + row * SCST;
    int idxs[16];
    #pragma unroll
    for (int t = 0; t < 16; ++t) {
        float bv = 3.0e38f; int bj = 0x7fffffff;
        for (int j = jlo; j < jlo + 96; ++j) {
            float v = srow[j];
            if (v < bv) { bv = v; bj = j; }
        }
        float ov = __shfl_xor(bv, 16);
        int   oj = __shfl_xor(bj, 16);
        int sel = (bv < ov || (bv == ov && bj < oj)) ? bj : oj;
        if (sel >= jlo && sel < jlo + 96) srow[sel] = 3.0e38f;   // owner marks used
        idxs[t] = sel;
    }

    // Aggregate: feats(row) = max_t elu(C_row + U_{idx_t})
    float cv[16], acc[16];
    {
        const float* cp = C + (size_t)(dstBase + row) * 16;
        #pragma unroll
        for (int d = 0; d < 16; ++d) { cv[d] = cp[d]; acc[d] = -3.0e38f; }
    }
    #pragma unroll
    for (int t = 0; t < 16; ++t) {
        const float* up = U + (size_t)(srcBase + idxs[t]) * 16;
        #pragma unroll
        for (int d = 0; d < 16; ++d)
            acc[d] = fmaxf(acc[d], eluf(cv[d] + up[d]));
    }
    if (grp == 0) {
        float* op = out + (size_t)(dstBase + row) * 16;
        #pragma unroll
        for (int d = 0; d < 16; ++d) op[d] = acc[d];
    }
}

// ---------------------------------------------------------------------------
// Mean pool over the 128 TS rows of each graph.
// ---------------------------------------------------------------------------
__global__ void pool_kernel(const float* __restrict__ f, float* __restrict__ pooled) {
    int g = blockIdx.x, d = threadIdx.x;     // blockDim == 16
    float s = 0.f;
    for (int r = 0; r < 128; ++r) s += f[(g * 128 + r) * 16 + d];
    pooled[g * 16 + d] = s * (1.0f / 128.0f);
}

// ---------------------------------------------------------------------------
// Head MLP 16->64->32->8->4->1 per graph; also emits batch_out = arange(64).
// ---------------------------------------------------------------------------
__global__ void head_kernel(const float* __restrict__ pooled,
                            const float* w1, const float* b1,
                            const float* w2, const float* b2,
                            const float* w3, const float* b3,
                            const float* w4, const float* b4,
                            const float* w5, const float* b5,
                            float* __restrict__ out) {
    __shared__ float s1[1024], sb1[64], s2[2048], sb2[32], s3[256], sb3[8],
                     s4[32], sb4[4], s5[4], sb5[1];
    int t = threadIdx.x;                     // blockDim == 64
    for (int i = t; i < 1024; i += 64) s1[i] = w1[i];
    for (int i = t; i < 2048; i += 64) s2[i] = w2[i];
    for (int i = t; i < 256; i += 64) s3[i] = w3[i];
    if (t < 64) sb1[t] = b1[t];
    if (t < 32) { sb2[t] = b2[t]; s4[t] = w4[t]; }
    if (t < 8)  sb3[t] = b3[t];
    if (t < 4)  { sb4[t] = b4[t]; s5[t] = w5[t]; }
    if (t == 0) sb5[0] = b5[0];
    __syncthreads();
    int g = t;
    float p[16];
    #pragma unroll
    for (int d = 0; d < 16; ++d) p[d] = pooled[g * 16 + d];
    float h1[64];
    for (int nn = 0; nn < 64; ++nn) {
        float s = sb1[nn];
        #pragma unroll
        for (int k = 0; k < 16; ++k) s += p[k] * s1[k * 64 + nn];
        h1[nn] = eluf(s);
    }
    float h2[32];
    for (int nn = 0; nn < 32; ++nn) {
        float s = sb2[nn];
        for (int k = 0; k < 64; ++k) s += h1[k] * s2[k * 32 + nn];
        h2[nn] = eluf(s);
    }
    float h3[8];
    for (int nn = 0; nn < 8; ++nn) {
        float s = sb3[nn];
        for (int k = 0; k < 32; ++k) s += h2[k] * s3[k * 8 + nn];
        h3[nn] = eluf(s);
    }
    float h4[4];
    for (int nn = 0; nn < 4; ++nn) {
        float s = sb4[nn];
        for (int k = 0; k < 8; ++k) s += h3[k] * s4[k * 4 + nn];
        h4[nn] = eluf(s);
    }
    float o = sb5[0];
    for (int k = 0; k < 4; ++k) o += h4[k] * s5[k];
    out[g] = o;
    ((int*)out)[64 + g] = g;                 // batch_out = arange(B), int32 bits
}

// ---------------------------------------------------------------------------
extern "C" void kernel_launch(void* const* d_in, const int* in_sizes, int n_in,
                              void* d_out, int out_size, void* d_ws, size_t ws_size,
                              hipStream_t stream) {
    (void)in_sizes; (void)n_in; (void)out_size; (void)ws_size;
    const float* x_ts   = (const float*)d_in[0];
    const float* x_lc   = (const float*)d_in[1];
    const float* ts_w1  = (const float*)d_in[4];
    const float* ts_b1  = (const float*)d_in[5];
    const float* ts_w2  = (const float*)d_in[6];
    const float* ts_b2  = (const float*)d_in[7];
    const float* lc_w1  = (const float*)d_in[8];
    const float* lc_b1  = (const float*)d_in[9];
    const float* lc_w2  = (const float*)d_in[10];
    const float* lc_b2  = (const float*)d_in[11];
    const float* conv_w = (const float*)d_in[12];
    const float* conv_b = (const float*)d_in[13];
    const float* out_w1 = (const float*)d_in[14];
    const float* out_b1 = (const float*)d_in[15];
    const float* out_w2 = (const float*)d_in[16];
    const float* out_b2 = (const float*)d_in[17];
    const float* out_w3 = (const float*)d_in[18];
    const float* out_b3 = (const float*)d_in[19];
    const float* out_w4 = (const float*)d_in[20];
    const float* out_b4 = (const float*)d_in[21];
    const float* out_w5 = (const float*)d_in[22];
    const float* out_b5 = (const float*)d_in[23];

    float* ws     = (float*)d_ws;
    float* ts_enc = ws;                  // 8192*16  f32
    float* lc_enc = ws + 131072;         // 12288*16 f32
    float* f1     = ws + 327680;         // 12288*16 f32
    float* f2     = ws + 524288;         // 8192*16  f32
    float* f3     = ws + 655360;         // 8192*16  f32
    float* Ubuf   = ws + 786432;         // 12288*16 f32
    float* Cbuf   = ws + 983040;         // 12288*16 f32
    float* nrm    = ws + 1179648;        // 12288    f32
    float* pooled = ws + 1191936;        // 64*16    f32
    _Float16* lc16 = (_Float16*)(ws + 1192960);   // 12288*16 f16
    _Float16* ts16 = (_Float16*)(ws + 1291264);   // 8192*16  f16
    _Float16* f116 = (_Float16*)(ws + 1356800);   // 12288*16 f16
    _Float16* f216 = (_Float16*)(ws + 1455104);   // 8192*16  f16

    float* outp = (float*)d_out;

    // Encoders
    enc_kernel<<<32, 256, 0, stream>>>(x_ts, 6, ts_w1, ts_b1, ts_w2, ts_b2, ts_enc, 8192);
    enc_kernel<<<48, 256, 0, stream>>>(x_lc, 5, lc_w1, lc_b1, lc_w2, lc_b2, lc_enc, 12288);

    // Conv1: src = dst = lc_enc
    prep_kernel<<<768, 32, 0, stream>>>(lc_enc, conv_w, conv_b, Ubuf, nrm, lc16, 0);
    prep_kernel<<<768, 32, 0, stream>>>(lc_enc, conv_w, conv_b, Cbuf, nrm, nullptr, 1);
    knn_kernel<<<768, 32, 0, stream>>>(lc16, lc16, Ubuf, Cbuf, nrm, f1, 192, 12);

    // Conv2: src = lc_enc (U, nrm, lc16 reused), dst = ts_enc
    prep_kernel<<<512, 32, 0, stream>>>(ts_enc, conv_w, conv_b, Cbuf, nrm, ts16, 1);
    knn_kernel<<<512, 32, 0, stream>>>(ts16, lc16, Ubuf, Cbuf, nrm, f2, 128, 8);

    // Conv3: src = f1, dst = f2
    prep_kernel<<<768, 32, 0, stream>>>(f1, conv_w, conv_b, Ubuf, nrm, f116, 0);
    prep_kernel<<<512, 32, 0, stream>>>(f2, conv_w, conv_b, Cbuf, nrm, f216, 1);
    knn_kernel<<<512, 32, 0, stream>>>(f216, f116, Ubuf, Cbuf, nrm, f3, 128, 8);

    // Pool + head
    pool_kernel<<<64, 16, 0, stream>>>(f3, pooled);
    head_kernel<<<1, 64, 0, stream>>>(pooled, out_w1, out_b1, out_w2, out_b2,
                                      out_w3, out_b3, out_w4, out_b4,
                                      out_w5, out_b5, outp);
}
